// ShapesReceiver_61546881351680
// MI455X (gfx1250) — matrix operand order repaired and checked by reading, not compiled, not run on hardware
//
#include <hip/hip_runtime.h>

// ---------------------------------------------------------------------------
// Problem constants (from reference): B,S,V,E,H
// ---------------------------------------------------------------------------
constexpr int Bc = 1024, Sc = 64, Vc = 2048, Ec = 512, Hc = 1024;

typedef __attribute__((ext_vector_type(16))) __bf16 v16bf;
typedef __attribute__((ext_vector_type(8)))  __bf16 v8bf;
typedef __attribute__((ext_vector_type(8)))  float  v8f;
typedef __attribute__((ext_vector_type(4)))  unsigned int v4u;
typedef int v4i_vs __attribute__((__vector_size__(16)));   // matches builtin param type

#define AS1 __attribute__((address_space(1)))
#define AS3 __attribute__((address_space(3)))

__device__ __forceinline__ unsigned short f2bf(float f) {
    unsigned int u = __float_as_uint(f);
    u += 0x7FFFu + ((u >> 16) & 1u);          // round-to-nearest-even
    return (unsigned short)(u >> 16);
}
__device__ __forceinline__ unsigned int pk2(float a, float b) {
    return (unsigned int)f2bf(a) | ((unsigned int)f2bf(b) << 16);
}
__device__ __forceinline__ float sigmoidf(float x) {
    return 1.0f / (1.0f + __expf(-x));
}

// ------------------------- CDNA5 async global->LDS -------------------------
// 16-byte async copy, tracked by ASYNCcnt; data does not transit VGPRs.
__device__ __forceinline__ void async_copy_b128(const void* gsrc, void* ldst) {
#if __has_builtin(__builtin_amdgcn_global_load_async_to_lds_b128)
    __builtin_amdgcn_global_load_async_to_lds_b128(
        (AS1 v4i_vs*)(uintptr_t)gsrc,
        (AS3 v4i_vs*)(unsigned int)(uintptr_t)ldst, 0, 0);
#else
    unsigned int  lo = (unsigned int)(uintptr_t)ldst;       // LDS offset (addr[31:0])
    unsigned long long ga = (unsigned long long)(uintptr_t)gsrc;
    asm volatile("global_load_async_to_lds_b128 %0, %1, off"
                 :: "v"(lo), "v"(ga) : "memory");
#endif
}
__device__ __forceinline__ void wait_async0() {
#if __has_builtin(__builtin_amdgcn_s_wait_asynccnt)
    __builtin_amdgcn_s_wait_asynccnt(0);
#else
    asm volatile("s_wait_asynccnt 0x0" ::: "memory");
#endif
}

__device__ __forceinline__ v16bf cat16(v8bf lo, v8bf hi) {
    return __builtin_shufflevector(lo, hi, 0,1,2,3,4,5,6,7,8,9,10,11,12,13,14,15);
}

// A-fragment (16x32 bf16, MxK) from an LDS panel laid out [rows][32] row-major.
// Per ISA 7.12.2: lane m=L&15, g=L>>4: elems 0..7 = K g*8..g*8+7, elems 8..15 = K 16+g*8..
__device__ __forceinline__ v16bf lda_frag(const unsigned short* ldsA, int mBase, int lane) {
    int m = mBase + (lane & 15);
    int g = lane >> 4;
    v8bf lo = *(const v8bf*)(ldsA + m * 32 + g * 8);
    v8bf hi = *(const v8bf*)(ldsA + m * 32 + 16 + g * 8);
    return cat16(lo, hi);
}
// B-fragment (32x16 bf16, KxN): lane n=L&15, g=L>>4: elems 0..7 = K g*8..g*8+7, elems 8..15 = K 16+g*8..
__device__ __forceinline__ v16bf ldb_frag(const unsigned short* ldsB, int nBase, int lane) {
    int n = nBase + (lane & 15);
    int g = lane >> 4;
    v8bf lo = *(const v8bf*)(ldsB + n * 32 + g * 8);
    v8bf hi = *(const v8bf*)(ldsB + n * 32 + 16 + g * 8);
    return cat16(lo, hi);
}

__device__ __forceinline__ v8f wmma_bf16(v16bf a, v16bf b, v8f c) {
    return __builtin_amdgcn_wmma_f32_16x16x32_bf16(false, a, false, b, (short)0, c, false, false);
}
__device__ __forceinline__ v8f zero_v8f() {
    v8f z;
    #pragma unroll
    for (int q = 0; q < 8; ++q) z[q] = 0.0f;
    return z;
}

// ---------------------------------------------------------------------------
// Small prep kernels (one-time, tiny vs. main GEMMs)
// ---------------------------------------------------------------------------
__global__ void cvt_bf16_kernel(const float* __restrict__ src,
                                unsigned short* __restrict__ dst, size_t n) {
    size_t i = (size_t)blockIdx.x * blockDim.x + threadIdx.x;
    size_t stride = (size_t)gridDim.x * blockDim.x;
    for (; i < n; i += stride) dst[i] = f2bf(src[i]);
}

// embT[e*V + v] = bf16(embedding[v*E + e])  (so B-operand rows are K-contiguous)
__global__ void cvt_transpose_kernel(const float* __restrict__ src,
                                     unsigned short* __restrict__ dst, int V, int E) {
    int i = blockIdx.x * blockDim.x + threadIdx.x;
    if (i >= V * E) return;
    int e = i / V, v = i - e * V;
    dst[i] = f2bf(src[(size_t)v * E + e]);
}

__global__ void bias_comb_kernel(const float* __restrict__ bih,
                                 const float* __restrict__ bhh,
                                 float* __restrict__ dst, int n) {
    int i = blockIdx.x * blockDim.x + threadIdx.x;
    if (i < n) dst[i] = bih[i] + bhh[i];
}

__global__ void state_init_kernel(float* __restrict__ c,
                                  unsigned short* __restrict__ h0, size_t n) {
    size_t i = (size_t)blockIdx.x * blockDim.x + threadIdx.x;
    size_t stride = (size_t)gridDim.x * blockDim.x;
    for (; i < n; i += stride) { c[i] = 0.0f; h0[i] = 0; }
}

// ---------------------------------------------------------------------------
// GEMM1: emb[M,N] = messages[M,K](f32, cvt on the fly) x embT[N,K]^T (bf16)
// M=65536, K=2048, N=512.  Block tile 128x256, BK=32, 8 waves each 64x64.
// A-operand must transit VGPRs (f32->bf16); B staged with async copies.
// ---------------------------------------------------------------------------
#define G1_BM 128
#define G1_BN 256
#define G1_BK 32

__global__ __launch_bounds__(256) void gemm1_kernel(
    const float* __restrict__ A, const unsigned short* __restrict__ Bt,
    float* __restrict__ Cf, unsigned short* __restrict__ Cbf,
    int M, int N, int K)
{
    __shared__ __align__(16) unsigned short ldsA[G1_BM * G1_BK];  // 8 KB
    __shared__ __align__(16) unsigned short ldsB[G1_BN * G1_BK];  // 16 KB

    const int tid = threadIdx.x;
    const int lane = tid & 31;
    const int wid = tid >> 5;
    const int wm = wid & 1;        // 2 row-groups of 64
    const int wn = wid >> 1;       // 4 col-groups of 64
    const int row0 = blockIdx.x * G1_BM;
    const int col0 = blockIdx.y * G1_BN;

    v8f acc[4][4];
    #pragma unroll
    for (int i = 0; i < 4; ++i)
        #pragma unroll
        for (int j = 0; j < 4; ++j) acc[i][j] = zero_v8f();

    const int am = tid >> 1;
    const int akh = (tid & 1) << 4;

    for (int k0 = 0; k0 < K; k0 += G1_BK) {
        // stage B: 256 cols x 32 K bf16 (64B per col) -- async, no VGPR transit
        {
            const unsigned short* src = Bt + (size_t)(col0 + tid) * K + k0;
            unsigned short* dst = ldsB + tid * G1_BK;
            async_copy_b128(src,      dst);
            async_copy_b128(src +  8, dst +  8);
            async_copy_b128(src + 16, dst + 16);
            async_copy_b128(src + 24, dst + 24);
        }
        // stage A: 128 rows x 32 K, f32 -> bf16
        {
            const float* src = A + (size_t)(row0 + am) * K + k0 + akh;
            __builtin_prefetch(src + G1_BK, 0, 0);
            const float4* s4 = (const float4*)src;
            float4 f0 = s4[0], f1 = s4[1], f2 = s4[2], f3 = s4[3];
            v4u lo, hi;
            lo[0] = pk2(f0.x, f0.y); lo[1] = pk2(f0.z, f0.w);
            lo[2] = pk2(f1.x, f1.y); lo[3] = pk2(f1.z, f1.w);
            hi[0] = pk2(f2.x, f2.y); hi[1] = pk2(f2.z, f2.w);
            hi[2] = pk2(f3.x, f3.y); hi[3] = pk2(f3.z, f3.w);
            v4u* dst = (v4u*)(ldsA + am * G1_BK + akh);
            dst[0] = lo; dst[1] = hi;
        }
        wait_async0();
        __syncthreads();

        v16bf af[4], bfr[4];
        #pragma unroll
        for (int i = 0; i < 4; ++i) af[i] = lda_frag(ldsA, wm * 64 + i * 16, lane);
        #pragma unroll
        for (int j = 0; j < 4; ++j) bfr[j] = ldb_frag(ldsB, wn * 64 + j * 16, lane);
        #pragma unroll
        for (int i = 0; i < 4; ++i)
            #pragma unroll
            for (int j = 0; j < 4; ++j)
                acc[i][j] = wmma_bf16(af[i], bfr[j], acc[i][j]);
        __syncthreads();
    }

    const int g = lane >> 4, nn = lane & 15;
    #pragma unroll
    for (int i = 0; i < 4; ++i) {
        #pragma unroll
        for (int j = 0; j < 4; ++j) {
            int col = col0 + wn * 64 + j * 16 + nn;
            #pragma unroll
            for (int r = 0; r < 8; ++r) {
                int row = row0 + wm * 64 + i * 16 + r + 8 * g;
                float v = acc[i][j][r];
                size_t idx = (size_t)row * N + col;
                Cf[idx] = v;
                Cbf[idx] = f2bf(v);
            }
        }
    }
}

// ---------------------------------------------------------------------------
// Fused LSTM step: gates[B,4H] = [emb_t | h_prev] @ [W_ih | W_hh]^T + bias,
// then elementwise cell update.  K = E + H = 1536.
// Double-buffered LDS with async global->LDS copies; ONE barrier per K-step.
// ---------------------------------------------------------------------------
#define LS_ROWS 128
#define LS_NU   32

__global__ __launch_bounds__(256) void lstm_step_kernel(
    const unsigned short* __restrict__ embBf,  // [B,S,E] bf16
    const unsigned short* __restrict__ Wih,    // [4H,E] bf16
    const unsigned short* __restrict__ Whh,    // [4H,H] bf16
    const float* __restrict__ bias,            // [4H] (b_ih + b_hh)
    const unsigned short* __restrict__ hPrev,  // [B,H] bf16
    unsigned short* __restrict__ hNext,        // [B,H] bf16
    float* __restrict__ cSt,                   // [B,H] f32 (in/out)
    int t)
{
    __shared__ __align__(16) unsigned short ldsA[2][LS_ROWS * 32];    // 2 x 8 KB
    __shared__ __align__(16) unsigned short ldsB[2][4 * LS_NU * 32];  // 2 x 8 KB

    const int tid = threadIdx.x;
    const int lane = tid & 31;
    const int wid = tid >> 5;
    const int row0 = blockIdx.x * LS_ROWS;
    const int n0 = blockIdx.y * LS_NU;

    v8f acc[4][2];
    #pragma unroll
    for (int gi = 0; gi < 4; ++gi) { acc[gi][0] = zero_v8f(); acc[gi][1] = zero_v8f(); }

    const int am = tid >> 1;
    const int akh = (tid & 1) << 4;
    const int br = tid >> 1;
    const int bg = br >> 5;
    const int bn = br & 31;

    auto stage = [&](int buf, int k0) {
        // A: [emb_t | h_prev] rows, 128 x 32 bf16
        {
            int kk = k0 + akh;
            const unsigned short* src = (kk < Ec)
                ? (embBf + ((size_t)(row0 + am) * Sc + t) * Ec + kk)
                : (hPrev + (size_t)(row0 + am) * Hc + (kk - Ec));
            unsigned short* dst = &ldsA[buf][am * 32 + akh];
            async_copy_b128(src,     dst);
            async_copy_b128(src + 8, dst + 8);
        }
        // B: 4 gates x 32 units, rows of [W_ih | W_hh]
        {
            int kk = k0 + akh;
            int wrow = bg * Hc + n0 + bn;
            const unsigned short* src = (kk < Ec)
                ? (Wih + (size_t)wrow * Ec + kk)
                : (Whh + (size_t)wrow * Hc + (kk - Ec));
            unsigned short* dst = &ldsB[buf][(bg * LS_NU + bn) * 32 + akh];
            async_copy_b128(src,     dst);
            async_copy_b128(src + 8, dst + 8);
        }
    };

    const int KT = (Ec + Hc) / 32;   // 48
    stage(0, 0);
    for (int it = 0; it < KT; ++it) {
        int buf = it & 1;
        wait_async0();      // own copies for ldsX[buf] landed
        __syncthreads();    // everyone's copies landed; prev readers of buf^1 done
        if (it + 1 < KT) stage(buf ^ 1, (it + 1) * 32);

        v16bf af = lda_frag(ldsA[buf], wid * 16, lane);
        #pragma unroll
        for (int gi = 0; gi < 4; ++gi) {
            #pragma unroll
            for (int j = 0; j < 2; ++j) {
                v16bf bfr = ldb_frag(&ldsB[buf][gi * LS_NU * 32], j * 16, lane);
                acc[gi][j] = wmma_bf16(af, bfr, acc[gi][j]);
            }
        }
    }

    // epilogue: bias + LSTM cell update
    const int gq = lane >> 4, nn = lane & 15;
    #pragma unroll
    for (int j = 0; j < 2; ++j) {
        int col = n0 + j * 16 + nn;
        float bi = bias[col];
        float bf_ = bias[Hc + col];
        float bg_ = bias[2 * Hc + col];
        float bo = bias[3 * Hc + col];
        #pragma unroll
        for (int r = 0; r < 8; ++r) {
            int row = row0 + wid * 16 + r + 8 * gq;
            size_t idx = (size_t)row * Hc + col;
            float iv = sigmoidf(acc[0][j][r] + bi);
            float fv = sigmoidf(acc[1][j][r] + bf_);
            float gv = tanhf(acc[2][j][r] + bg_);
            float ov = sigmoidf(acc[3][j][r] + bo);
            float cn = fv * cSt[idx] + iv * gv;
            cSt[idx] = cn;
            hNext[idx] = f2bf(ov * tanhf(cn));
        }
    }
}

// ---------------------------------------------------------------------------
// Final projection: out[B,H] = h @ out_W^T + out_b.  Tile 128x128, 8 waves
// 64x32, async double-buffered staging like the LSTM step.
// ---------------------------------------------------------------------------
__global__ __launch_bounds__(256) void proj_kernel(
    const unsigned short* __restrict__ hBf,  // [B,H] bf16
    const unsigned short* __restrict__ Wo,   // [H,H] bf16 (row n -> out col n)
    const float* __restrict__ ob,
    float* __restrict__ out, int M, int N, int K)
{
    __shared__ __align__(16) unsigned short ldsA[2][128 * 32];  // 2 x 8 KB
    __shared__ __align__(16) unsigned short ldsB[2][128 * 32];  // 2 x 8 KB

    const int tid = threadIdx.x;
    const int lane = tid & 31;
    const int wid = tid >> 5;
    const int wm = wid & 1;
    const int wn = wid >> 1;
    const int row0 = blockIdx.x * 128;
    const int col0 = blockIdx.y * 128;

    v8f acc[4][2];
    #pragma unroll
    for (int i = 0; i < 4; ++i) { acc[i][0] = zero_v8f(); acc[i][1] = zero_v8f(); }

    const int am = tid >> 1;
    const int akh = (tid & 1) << 4;

    auto stage = [&](int buf, int k0) {
        {
            const unsigned short* src = hBf + (size_t)(row0 + am) * K + k0 + akh;
            unsigned short* dst = &ldsA[buf][am * 32 + akh];
            async_copy_b128(src,     dst);
            async_copy_b128(src + 8, dst + 8);
        }
        {
            const unsigned short* src = Wo + (size_t)(col0 + am) * K + k0 + akh;
            unsigned short* dst = &ldsB[buf][am * 32 + akh];
            async_copy_b128(src,     dst);
            async_copy_b128(src + 8, dst + 8);
        }
    };

    const int KT = K / 32;
    stage(0, 0);
    for (int it = 0; it < KT; ++it) {
        int buf = it & 1;
        wait_async0();
        __syncthreads();
        if (it + 1 < KT) stage(buf ^ 1, (it + 1) * 32);

        v16bf af[4];
        #pragma unroll
        for (int i = 0; i < 4; ++i) af[i] = lda_frag(ldsA[buf], wm * 64 + i * 16, lane);
        #pragma unroll
        for (int j = 0; j < 2; ++j) {
            v16bf bfr = ldb_frag(ldsB[buf], wn * 32 + j * 16, lane);
            #pragma unroll
            for (int i = 0; i < 4; ++i)
                acc[i][j] = wmma_bf16(af[i], bfr, acc[i][j]);
        }
    }

    const int g = lane >> 4, nn = lane & 15;
    #pragma unroll
    for (int i = 0; i < 4; ++i) {
        #pragma unroll
        for (int j = 0; j < 2; ++j) {
            int col = col0 + wn * 32 + j * 16 + nn;
            float bb = ob[col];
            #pragma unroll
            for (int r = 0; r < 8; ++r) {
                int row = row0 + wm * 64 + i * 16 + r + 8 * g;
                out[(size_t)row * N + col] = acc[i][j][r] + bb;
            }
        }
    }
}

// ---------------------------------------------------------------------------
// Host-side orchestration
// ---------------------------------------------------------------------------
extern "C" void kernel_launch(void* const* d_in, const int* in_sizes, int n_in,
                              void* d_out, int out_size, void* d_ws, size_t ws_size,
                              hipStream_t stream)
{
    const float* messages  = (const float*)d_in[0];  // [B,S,V]
    const float* embedding = (const float*)d_in[1];  // [V,E]
    const float* W_ih      = (const float*)d_in[2];  // [4H,E]
    const float* W_hh      = (const float*)d_in[3];  // [4H,H]
    const float* b_ih      = (const float*)d_in[4];  // [4H]
    const float* b_hh      = (const float*)d_in[5];  // [4H]
    const float* out_W     = (const float*)d_in[6];  // [H,H]
    const float* out_b     = (const float*)d_in[7];  // [H]

    float* outP = (float*)d_out;                       // [B,H]
    float* embF = outP + (size_t)Bc * Hc;              // [B,S,E] (output #2)

    char* ws = (char*)d_ws;
    size_t off = 0;
    auto take = [&](size_t bytes) -> char* {
        char* p = ws + off;
        off += (bytes + 255) & ~(size_t)255;
        return p;
    };
    unsigned short* embT  = (unsigned short*)take((size_t)Ec * Vc * 2);        // [E,V]
    unsigned short* WihB  = (unsigned short*)take((size_t)4 * Hc * Ec * 2);    // [4H,E]
    unsigned short* WhhB  = (unsigned short*)take((size_t)4 * Hc * Hc * 2);    // [4H,H]
    unsigned short* WoB   = (unsigned short*)take((size_t)Hc * Hc * 2);        // [H,H]
    float*          biasC = (float*)take((size_t)4 * Hc * 4);                  // [4H]
    unsigned short* embBf = (unsigned short*)take((size_t)Bc * Sc * Ec * 2);   // [B,S,E]
    unsigned short* hA    = (unsigned short*)take((size_t)Bc * Hc * 2);        // h ping
    unsigned short* hB    = (unsigned short*)take((size_t)Bc * Hc * 2);        // h pong
    float*          cBuf  = (float*)take((size_t)Bc * Hc * 4);                 // c state

    // 1) precision prep (bf16 weights, transposed embedding, combined bias)
    cvt_transpose_kernel<<<(Vc * Ec + 255) / 256, 256, 0, stream>>>(embedding, embT, Vc, Ec);
    cvt_bf16_kernel<<<2048, 256, 0, stream>>>(W_ih, WihB, (size_t)4 * Hc * Ec);
    cvt_bf16_kernel<<<2048, 256, 0, stream>>>(W_hh, WhhB, (size_t)4 * Hc * Hc);
    cvt_bf16_kernel<<<1024, 256, 0, stream>>>(out_W, WoB, (size_t)Hc * Hc);
    bias_comb_kernel<<<(4 * Hc + 255) / 256, 256, 0, stream>>>(b_ih, b_hh, biasC, 4 * Hc);
    state_init_kernel<<<1024, 256, 0, stream>>>(cBuf, hA, (size_t)Bc * Hc);

    // 2) soft-embedding GEMM: [65536,2048] x [2048,512]
    gemm1_kernel<<<dim3((Bc * Sc) / G1_BM, Ec / G1_BN), 256, 0, stream>>>(
        messages, embT, embF, embBf, Bc * Sc, Ec, Vc);

    // 3) 64 fused LSTM steps (ping-pong h)
    for (int t = 0; t < Sc; ++t) {
        const unsigned short* hp = (t & 1) ? hB : hA;
        unsigned short*       hn = (t & 1) ? hA : hB;
        lstm_step_kernel<<<dim3(Bc / LS_ROWS, Hc / LS_NU), 256, 0, stream>>>(
            embBf, WihB, WhhB, biasC, hp, hn, cBuf, t);
    }
    // t=63 (odd) wrote hA -> final hidden state is hA

    // 4) output projection
    proj_kernel<<<dim3(Bc / 128, Hc / 128), 256, 0, stream>>>(
        hA, WoB, out_b, outP, Bc, Hc, Hc);
}
